// DifferentiableAttention_9818295239474
// MI455X (gfx1250) — matrix operand order repaired and checked by reading, not compiled, not run on hardware
//
#include <hip/hip_runtime.h>

// ---------------------------------------------------------------------------
// Types for CDNA5 WMMA (wave32): v_wmma_f32_16x16x32_bf16
// ---------------------------------------------------------------------------
typedef __attribute__((ext_vector_type(16))) __bf16 v16bf;
typedef __attribute__((ext_vector_type(8)))  float  v8f;
typedef __attribute__((ext_vector_type(4)))  unsigned int u32x4;

#define LAMBDA_INIT 0.3555090687f   // 0.8 - 0.6*exp(-0.3)

__device__ __forceinline__ __bf16 f2bf(float f) { return (__bf16)f; }  // native cvt

union FragU { struct { u32x4 lo, hi; } p; v16bf v; };
__device__ __forceinline__ v16bf make_frag(u32x4 lo, u32x4 hi) {
  FragU u; u.p.lo = lo; u.p.hi = hi; return u.v;
}

// A-fragment (16x32 bf16, ISA interleaved halves): lane<16 -> K0..7,K16..23
__device__ __forceinline__ v16bf lds_fragA(const __bf16* row, int half) {
  u32x4 lo = *(const u32x4*)(row + half * 8);
  u32x4 hi = *(const u32x4*)(row + 16 + half * 8);
  return make_frag(lo, hi);
}
// B-fragment (32x16 bf16, ISA interleaved halves): lane<16 -> K0..7,K16..23
__device__ __forceinline__ v16bf lds_fragB(const __bf16* col, int half) {
  const __bf16* p = col + half * 8;
  return make_frag(*(const u32x4*)p, *(const u32x4*)(p + 16));
}

__device__ __forceinline__ v8f wmma_bf16(v16bf a, v16bf b, v8f c) {
  return __builtin_amdgcn_wmma_f32_16x16x32_bf16(false, a, false, b, (short)0, c,
                                                 false, false);
}

// ---------------------------------------------------------------------------
// lambda = exp(dot(lq1,lk1)) - exp(dot(lq2,lk2)) + LAMBDA_INIT   (G = 512)
// ---------------------------------------------------------------------------
__global__ __launch_bounds__(256)
void lambda_kernel(const float* lq1, const float* lk1,
                   const float* lq2, const float* lk2, float* lamOut) {
  __shared__ float red[2][256];
  int tid = threadIdx.x;
  float s1 = 0.f, s2 = 0.f;
  for (int i = tid; i < 512; i += 256) { s1 += lq1[i] * lk1[i]; s2 += lq2[i] * lk2[i]; }
  red[0][tid] = s1; red[1][tid] = s2; __syncthreads();
  for (int off = 128; off > 0; off >>= 1) {
    if (tid < off) { red[0][tid] += red[0][tid + off]; red[1][tid] += red[1][tid + off]; }
    __syncthreads();
  }
  if (tid == 0) lamOut[0] = __expf(red[0][0]) - __expf(red[1][0]) + LAMBDA_INIT;
}

// ---------------------------------------------------------------------------
// Generic tiled WMMA GEMM:  C[M,N] = A[M,K] * B[K,N]
//   AMODE 0: A f32 row-major (+batch stride)      -> convert to bf16 on load
//   AMODE 1: A bf16 attn-gather: A[s][c] = attn[((b*16 + c/64)*1024 + s)*64 + c%64]
//   AMODE 2: A bf16 row-major (+batch stride)
//   OMODE 0: f32 row-major   OMODE 1: bf16 row-major   OMODE 2: bf16 transposed
// B always f32 row-major [K][N], shared across batch. Dims multiples of tiles.
// 256 threads = 8 waves; 128x256 block; waves in 2(M) x 4(N), 4x4 sixteens each
// => 16 WMMAs per K-step per wave from 8 fragment loads.
// ---------------------------------------------------------------------------
#define GBM 128
#define GBN 256
#define GBK 32
#define GLS 40   // LDS row stride (bf16 elems), 80B = multiple of 16B

template <int AMODE, int OMODE>
__global__ __launch_bounds__(256)
void gemm_wmma(const void* __restrict__ Aptr, const float* __restrict__ Bptr,
               void* __restrict__ Outp, int M, int N, int K,
               long strideA, long strideO) {
  __shared__ __bf16 Alds[GBM * GLS];
  __shared__ __bf16 Blds[GBN * GLS];   // transposed: [n][k]

  const int bz = blockIdx.z;
  const int m0 = blockIdx.y * GBM;
  const int n0 = blockIdx.x * GBN;
  const int tid = threadIdx.x;
  const int wave = tid >> 5, lane = tid & 31;
  const int half = lane >> 4, r = lane & 15;
  const int wm = wave >> 2;   // 0..1 -> rows wm*64
  const int wn = wave & 3;    // 0..3 -> cols wn*64

  v8f acc[4][4];
#pragma unroll
  for (int i = 0; i < 4; i++)
#pragma unroll
    for (int j = 0; j < 4; j++) acc[i][j] = (v8f)0.f;

  for (int k0 = 0; k0 < K; k0 += GBK) {
    __syncthreads();
    // ---- stage A tile (GBM x GBK) : 512 chunks of 8 elems, 2 per thread
    for (int c = tid; c < (GBM * GBK) / 8; c += 256) {
      int row = c >> 2, kk = (c & 3) * 8;
      int gm = m0 + row, gk = k0 + kk;
      __attribute__((aligned(16))) __bf16 tmp[8];
      if (AMODE == 0) {
        const float* src = (const float*)Aptr + (long)bz * strideA + (long)gm * K + gk;
        if (gk + GBK < K) __builtin_prefetch(src + GBK, 0, 0);
#pragma unroll
        for (int i = 0; i < 8; i++) tmp[i] = f2bf(src[i]);
      } else if (AMODE == 1) {
        const __bf16* at = (const __bf16*)Aptr;
        const __bf16* src = at + (((long)bz * 16 + (gk >> 6)) * 1024 + gm) * 64 + (gk & 63);
        if (gk + GBK < K) __builtin_prefetch(src + 1024 * 64 / 2, 0, 0);  // half a head ahead
        *(u32x4*)tmp = *(const u32x4*)src;
      } else {
        const __bf16* src = (const __bf16*)Aptr + (long)bz * strideA + (long)gm * K + gk;
        if (gk + GBK < K) __builtin_prefetch(src + GBK, 0, 0);
        *(u32x4*)tmp = *(const u32x4*)src;
      }
      *(u32x4*)&Alds[row * GLS + kk] = *(u32x4*)tmp;
    }
    // ---- stage B tile (GBK x GBN f32) transposed into Blds[n][k], 4 per thread
    for (int c = tid; c < (GBK * GBN) / 8; c += 256) {
      int kk = c >> 5, nn = (c & 31) * 8;
      const float* src = Bptr + (long)(k0 + kk) * N + n0 + nn;
      if (k0 + GBK < K) __builtin_prefetch(src + (long)GBK * N, 0, 0);
#pragma unroll
      for (int i = 0; i < 8; i++) Blds[(nn + i) * GLS + kk] = f2bf(src[i]);
    }
    __syncthreads();
    // ---- compute: 16 WMMAs from 4 A-frags + 4 B-frags
    v16bf af[4], bf[4];
#pragma unroll
    for (int i = 0; i < 4; i++) af[i] = lds_fragA(&Alds[(wm * 64 + i * 16 + r) * GLS], half);
#pragma unroll
    for (int j = 0; j < 4; j++) bf[j] = lds_fragB(&Blds[(wn * 64 + j * 16 + r) * GLS], half);
#pragma unroll
    for (int i = 0; i < 4; i++)
#pragma unroll
      for (int j = 0; j < 4; j++) acc[i][j] = wmma_bf16(af[i], bf[j], acc[i][j]);
  }

  // ---- store (C layout: element e -> row j = e + 8*half, col = r)
#pragma unroll
  for (int i = 0; i < 4; i++)
#pragma unroll
    for (int j = 0; j < 4; j++)
#pragma unroll
      for (int e = 0; e < 8; e++) {
        int gm = m0 + wm * 64 + i * 16 + e + half * 8;
        int gn = n0 + wn * 64 + j * 16 + r;
        float v = acc[i][j][e];
        if (OMODE == 0) {
          ((float*)Outp)[(long)bz * strideO + (long)gm * N + gn] = v;
        } else if (OMODE == 1) {
          ((__bf16*)Outp)[(long)bz * strideO + (long)gm * N + gn] = f2bf(v);
        } else {
          ((__bf16*)Outp)[(long)bz * strideO + (long)gn * M + gm] = f2bf(v);  // transposed
        }
      }
}

// ---------------------------------------------------------------------------
// Flash-style causal differential attention.
// Grid (S/64, 16 pairs, B), block = 128 (4 waves). Each wave owns 16 Q rows.
// qkv: bf16 (B, S, 3072). attn out: bf16 (B, 16, S, 64), scaled by (1-l_init).
// The causal mask is applied ONLY in the diagonal block (templated), so the
// main KV loop carries no per-element compares.
// ---------------------------------------------------------------------------
template <bool MASKED>
__device__ __forceinline__ void online_update(v8f (&s)[4], v8f (&o)[4],
                                              float (&mst)[8], float (&lst)[8],
                                              int row_base, int kv0, int r, int half) {
#pragma unroll
  for (int e = 0; e < 8; e++) {
    float vals[4];
    float mx = -3e38f;
#pragma unroll
    for (int t = 0; t < 4; t++) {
      float v = s[t][e] * 0.125f;               // 1/sqrt(64)
      if (MASKED) {
        int row = row_base + e + 8 * half;
        int col = kv0 + t * 16 + r;
        if (col > row) v = -3e38f;              // causal mask (diagonal block only)
      }
      vals[t] = v;
      mx = fmaxf(mx, v);
    }
#pragma unroll
    for (int m = 1; m < 16; m <<= 1) mx = fmaxf(mx, __shfl_xor(mx, m, 16));
    float mnew = fmaxf(mst[e], mx);
    float corr = __expf(mst[e] - mnew);
    float rsum = 0.f;
#pragma unroll
    for (int t = 0; t < 4; t++) { vals[t] = __expf(vals[t] - mnew); rsum += vals[t]; }
#pragma unroll
    for (int m = 1; m < 16; m <<= 1) rsum += __shfl_xor(rsum, m, 16);
    lst[e] = lst[e] * corr + rsum;
    mst[e] = mnew;
#pragma unroll
    for (int t = 0; t < 4; t++) { o[t][e] *= corr; s[t][e] = vals[t]; }
  }
}

template <bool MASKED>
__device__ __forceinline__ void attn_block(
    const __bf16* __restrict__ qkv, __bf16* Klds, __bf16* Vlds,
    __bf16* Plds0, __bf16* Plds1,
    int b, int hk, int kv0, int row_base, int tid, int wave, int half, int r,
    bool prefetch_next,
    const v16bf (&q1)[2], const v16bf (&q2)[2],
    v8f (&o1)[4], v8f (&o2)[4],
    float (&m1)[8], float (&l1)[8], float (&m2)[8], float (&l2)[8]) {
  __syncthreads();
  // ---- stage K tile [kv][hd]: k[b,hk,s,d] = qkv[b][hk*128+s/8][2048+(s%8)*64+d]
  for (int c = tid; c < 64 * 8; c += 128) {
    int sl = c >> 3, d0 = (c & 7) * 8;
    int sg = kv0 + sl;
    long src = ((long)(b * 1024 + hk * 128 + (sg >> 3))) * 3072 + 2048 + ((sg & 7) * 64) + d0;
    if (prefetch_next) __builtin_prefetch(qkv + src + 8L * 3072, 0, 0);  // next kv block
    *(u32x4*)&Klds[sl * 72 + d0] = *(const u32x4*)(qkv + src);
  }
  // ---- stage V tile transposed [hd][kv]: v channel base 2560
  for (int c = tid; c < 64 * 8; c += 128) {
    int sl = c >> 3, d0 = (c & 7) * 8;
    int sg = kv0 + sl;
    long src = ((long)(b * 1024 + hk * 128 + (sg >> 3))) * 3072 + 2560 + ((sg & 7) * 64) + d0;
    if (prefetch_next) __builtin_prefetch(qkv + src + 8L * 3072, 0, 0);
    __attribute__((aligned(16))) __bf16 tmp[8];
    *(u32x4*)tmp = *(const u32x4*)(qkv + src);
#pragma unroll
    for (int i = 0; i < 8; i++) Vlds[(d0 + i) * 72 + sl] = tmp[i];
  }
  __syncthreads();

  // ---- scores: S = Q K^T   (B-frag cols = kv, K-dim = hd)
  v8f s1[4], s2[4];
#pragma unroll
  for (int t = 0; t < 4; t++) { s1[t] = (v8f)0.f; s2[t] = (v8f)0.f; }
#pragma unroll
  for (int t = 0; t < 4; t++) {
    v16bf kb0 = lds_fragB(&Klds[(t * 16 + r) * 72 + 0], half);
    v16bf kb1 = lds_fragB(&Klds[(t * 16 + r) * 72 + 32], half);
    s1[t] = wmma_bf16(q1[0], kb0, s1[t]);
    s1[t] = wmma_bf16(q1[1], kb1, s1[t]);
    s2[t] = wmma_bf16(q2[0], kb0, s2[t]);
    s2[t] = wmma_bf16(q2[1], kb1, s2[t]);
  }

  online_update<MASKED>(s1, o1, m1, l1, row_base, kv0, r, half);
  online_update<MASKED>(s2, o2, m2, l2, row_base, kv0, r, half);

  // ---- P -> LDS (bf16), then P @ V
#pragma unroll
  for (int t = 0; t < 4; t++)
#pragma unroll
    for (int e = 0; e < 8; e++) {
      int pr = wave * 16 + e + 8 * half, pc = t * 16 + r;
      Plds0[pr * 72 + pc] = f2bf(s1[t][e]);
      Plds1[pr * 72 + pc] = f2bf(s2[t][e]);
    }
  __syncthreads();
#pragma unroll
  for (int f = 0; f < 2; f++) {     // K-dim (kv) halves 0..31 / 32..63
    v16bf p1 = lds_fragA(&Plds0[(wave * 16 + r) * 72 + f * 32], half);
    v16bf p2 = lds_fragA(&Plds1[(wave * 16 + r) * 72 + f * 32], half);
#pragma unroll
    for (int t = 0; t < 4; t++) {   // hd tiles
      v16bf vb = lds_fragB(&Vlds[(t * 16 + r) * 72 + f * 32], half);
      o1[t] = wmma_bf16(p1, vb, o1[t]);
      o2[t] = wmma_bf16(p2, vb, o2[t]);
    }
  }
}

__global__ __launch_bounds__(128)
void diff_attn(const __bf16* __restrict__ qkv, const float* __restrict__ lamBuf,
               __bf16* __restrict__ attn) {
  __shared__ __bf16 Klds[64 * 72];      // [kv][hd]
  __shared__ __bf16 Vlds[64 * 72];      // [hd][kv]  (transposed)
  __shared__ __bf16 Plds[2][64 * 72];   // [map][row][col]

  const int b = blockIdx.z, h = blockIdx.y, rb = blockIdx.x;
  const int row0 = rb * 64;
  const int tid = threadIdx.x;
  const int wave = tid >> 5, lane = tid & 31, half = lane >> 4, r = lane & 15;
  const int row_base = row0 + wave * 16;
  const float lam = lamBuf[0];
  const int hk = h >> 1;

  // ---- Q fragments (A-layout) loaded straight from global with head scramble:
  // q[b,hq,s,d] = qkv[b][hq*32 + s/32][(s%32)*64 + d]
  const int s_row = row_base + r;
  v16bf q1[2], q2[2];
  {
    long b1 = ((long)(b * 1024 + (2 * h) * 32 + (s_row >> 5))) * 3072 + ((s_row & 31) * 64);
    long b2 = ((long)(b * 1024 + (2 * h + 1) * 32 + (s_row >> 5))) * 3072 + ((s_row & 31) * 64);
#pragma unroll
    for (int f = 0; f < 2; f++) {
      int d0 = f * 32;
      q1[f] = make_frag(*(const u32x4*)(qkv + b1 + d0 + half * 8),
                        *(const u32x4*)(qkv + b1 + d0 + 16 + half * 8));
      q2[f] = make_frag(*(const u32x4*)(qkv + b2 + d0 + half * 8),
                        *(const u32x4*)(qkv + b2 + d0 + 16 + half * 8));
    }
  }

  float m1[8], l1[8], m2[8], l2[8];
  v8f o1[4], o2[4];
#pragma unroll
  for (int e = 0; e < 8; e++) { m1[e] = -3e38f; m2[e] = -3e38f; l1[e] = 0.f; l2[e] = 0.f; }
#pragma unroll
  for (int t = 0; t < 4; t++) { o1[t] = (v8f)0.f; o2[t] = (v8f)0.f; }

  // Unmasked KV blocks strictly below the diagonal
  for (int blk = 0; blk < rb; ++blk)
    attn_block<false>(qkv, Klds, Vlds, Plds[0], Plds[1], b, hk, blk * 64, row_base,
                      tid, wave, half, r, /*prefetch*/ true,
                      q1, q2, o1, o2, m1, l1, m2, l2);
  // Diagonal block with causal mask
  attn_block<true>(qkv, Klds, Vlds, Plds[0], Plds[1], b, hk, rb * 64, row_base,
                   tid, wave, half, r, /*prefetch*/ false,
                   q1, q2, o1, o2, m1, l1, m2, l2);

  // ---- finalize: (o1/l1 - lam*o2/l2) * (1 - LAMBDA_INIT) -> attn (B,16,S,64)
#pragma unroll
  for (int t = 0; t < 4; t++)
#pragma unroll
    for (int e = 0; e < 8; e++) {
      int row = row_base + e + 8 * half;
      int d = t * 16 + r;
      float v = (o1[t][e] / l1[e] - lam * o2[t][e] / l2[e]) * (1.0f - LAMBDA_INIT);
      attn[(((long)b * 16 + h) * 1024 + row) * 64 + d] = f2bf(v);
    }
}

// ---------------------------------------------------------------------------
// Host-side orchestration
// ---------------------------------------------------------------------------
extern "C" void kernel_launch(void* const* d_in, const int* in_sizes, int n_in,
                              void* d_out, int out_size, void* d_ws, size_t ws_size,
                              hipStream_t stream) {
  const float* x     = (const float*)d_in[0];   // (4,1024,1024)
  const float* Wqkv  = (const float*)d_in[1];   // (1024,3072)
  const float* Wproj = (const float*)d_in[2];   // (1024,1024)
  const float* lq1   = (const float*)d_in[3];
  const float* lk1   = (const float*)d_in[4];
  const float* lq2   = (const float*)d_in[5];
  const float* lk2   = (const float*)d_in[6];

  char* ws = (char*)d_ws;
  float*  lam  = (float*)ws;                          // 256 B
  __bf16* attn = (__bf16*)(ws + 256);                 // 4*16*1024*64 bf16 = 8 MB
  __bf16* qkv  = (__bf16*)(ws + 256 + (8u << 20));    // 4096*3072 bf16 = 24 MB
  __bf16* y1T  = qkv;                                 // reuse qkv region after attention

  // 1) lambda scalar
  lambda_kernel<<<dim3(1), dim3(256), 0, stream>>>(lq1, lk1, lq2, lk2, lam);

  // 2) qkv = x @ Wqkv    (M=4096, N=3072, K=1024)  f32 in -> bf16 out
  gemm_wmma<0, 1><<<dim3(3072 / GBN, 4096 / GBM, 1), dim3(256), 0, stream>>>(
      (const void*)x, Wqkv, (void*)qkv, 4096, 3072, 1024, 0L, 0L);

  // 3) differential flash attention -> attn (B,16,S,64) bf16
  diff_attn<<<dim3(1024 / 64, 16, 4), dim3(128), 0, stream>>>(qkv, lam, attn);

  // 4) y1T[b] = (attn_rows @ Wproj)^T   (per batch 1024^3), transposed bf16 store
  gemm_wmma<1, 2><<<dim3(1024 / GBN, 1024 / GBM, 4), dim3(256), 0, stream>>>(
      (const void*)attn, Wproj, (void*)y1T, 1024, 1024, 1024, 0L, 1024L * 1024L);

  // 5) out[b] = y1T[b] @ Wproj  -> f32 d_out
  gemm_wmma<2, 0><<<dim3(1024 / GBN, 1024 / GBM, 4), dim3(256), 0, stream>>>(
      (const void*)y1T, Wproj, d_out, 1024, 1024, 1024, 1024L * 1024L, 1024L * 1024L);
}
